// RNNDecoder_76931454206280
// MI455X (gfx1250) — matrix operand order repaired and checked by reading, not compiled, not run on hardware
//
#include <hip/hip_runtime.h>
#include <math.h>

// ---------------------------------------------------------------------------
// Problem constants (from reference): VOCAB=32000, EMB=512, H=1024, P=640,
// B=16, T=512. ROWS = B*T = 8192, 4H = 4096.
// ---------------------------------------------------------------------------
#define EMB   512
#define HID   1024
#define FOURH 4096
#define PDIM  640
#define BSZ   16
#define TLEN  512
#define ROWS  8192
#define NWG   32          // workgroups in the persistent recurrent kernel

// padded LDS row length (elements): 1032*2 = 2064 B => lane stride 516 dwords
// == 4 mod 64 banks => lanes 0..15 hit banks 0,4,...,60: conflict-free b128.
#define UPADE 1032
#define ULDSE (128 * UPADE)   // U slice: 128 rows x 1032 bf16
#define HLDSE (BSZ * UPADE)   // h stage: 16 rows x 1032 bf16
#define DYNLDS_BYTES ((size_t)(ULDSE + HLDSE) * 2)

typedef __attribute__((ext_vector_type(16))) __bf16 v16bf;
typedef __attribute__((ext_vector_type(8)))  float  v8f;

union FragBF {
    v16bf    v;
    unsigned u[8];
    uint4    q[2];
};

__device__ __forceinline__ unsigned short f2bf(float f) {
    unsigned u = __float_as_uint(f);
    unsigned r = (u + 0x7FFFu + ((u >> 16) & 1u)) >> 16;   // RNE
    return (unsigned short)r;
}

__device__ __forceinline__ float sigmoidf_(float x) {
    return 1.0f / (1.0f + __expf(-x));
}

// Software grid barrier across the NWG persistent workgroups, with explicit
// agent-scope release/acquire fences (emits global_wb / global_inv) so that
// plain stores to the h double-buffer are visible across WGPs each step.
__device__ __forceinline__ void grid_barrier(unsigned* bar, unsigned target) {
    __syncthreads();
    __builtin_amdgcn_fence(__ATOMIC_RELEASE, "agent");
    if (threadIdx.x == 0) {
        __hip_atomic_fetch_add(bar, 1u, __ATOMIC_ACQ_REL, __HIP_MEMORY_SCOPE_AGENT);
        while (__hip_atomic_load(bar, __ATOMIC_ACQUIRE, __HIP_MEMORY_SCOPE_AGENT) < target) {
            __builtin_amdgcn_s_sleep(1);
        }
    }
    __syncthreads();
    __builtin_amdgcn_fence(__ATOMIC_ACQUIRE, "agent");
}

// ---------------------------------------------------------------------------
// Utility kernels
// ---------------------------------------------------------------------------
__global__ void zero_u32(unsigned* p, int n) {
    int i = blockIdx.x * blockDim.x + threadIdx.x;
    if (i < n) p[i] = 0u;
}

// out[n*K + k] = bf16(in[k*N + n])   (weight transpose+convert, [K][N] -> [N][K])
__global__ void transpose_f32_bf16(const float* __restrict__ in,
                                   unsigned short* __restrict__ out,
                                   int K, int N) {
    size_t total = (size_t)K * (size_t)N;
    for (size_t idx = (size_t)blockIdx.x * blockDim.x + threadIdx.x; idx < total;
         idx += (size_t)gridDim.x * blockDim.x) {
        size_t n = idx / (size_t)K;
        size_t k = idx % (size_t)K;
        out[idx] = f2bf(in[k * (size_t)N + n]);
    }
}

__global__ void cvt_f32_bf16(const float* __restrict__ in,
                             unsigned short* __restrict__ out, size_t total) {
    for (size_t idx = (size_t)blockIdx.x * blockDim.x + threadIdx.x; idx < total;
         idx += (size_t)gridDim.x * blockDim.x) {
        out[idx] = f2bf(in[idx]);
    }
}

// xb[row][e] = bf16(embed[tokens[row]][e]); row = b*T + t
__global__ void gather_embed(const int* __restrict__ tokens,
                             const float* __restrict__ embed,
                             unsigned short* __restrict__ xb) {
    const int row = blockIdx.x;
    const int tok = tokens[row];
    const float* src = embed + (size_t)tok * EMB;
    unsigned short* dst = xb + (size_t)row * EMB;
    for (int j = threadIdx.x; j < EMB; j += blockDim.x) dst[j] = f2bf(src[j]);
}

// ---------------------------------------------------------------------------
// Generic bf16 WMMA GEMM:  C[M][N] = A[M][K] * B[K][N] + bias[N]
// A row-major bf16, B pre-transposed as Bt[N][K] bf16, C fp32.
// Block = 256 threads (8 waves); each wave owns one 16x16 output tile;
// block covers 16 rows x 128 cols.  K iterated in 32-wide WMMA steps.
// Fragment k order: lane half h holds k = 16h..16h+15 (contiguous) in both A and B.
// ---------------------------------------------------------------------------
__global__ __launch_bounds__(256) void gemm_bf16_wmma(
    const unsigned short* __restrict__ A,
    const unsigned short* __restrict__ Bt,
    const float* __restrict__ bias,
    float* __restrict__ C,
    int M, int N, int K) {
    const int mt   = blockIdx.x;
    const int wave = threadIdx.x >> 5;
    const int lane = threadIdx.x & 31;
    const int n0   = blockIdx.y * 128 + wave * 16;
    if (n0 >= N) return;   // never taken with exact grids; keeps EXEC full

    const int mrow = lane & 15;
    const int kb   = (lane < 16) ? 0 : 8;
    const int kg   = (lane < 16) ? 0 : 16;   // A,B: lanes 16-31 hold K=16..31
    const unsigned short* arow = A + (size_t)(mt * 16 + mrow) * K;
    const unsigned short* brow = Bt + (size_t)(n0 + (lane & 15)) * K + kg;

    v8f acc = {};
    for (int k0 = 0; k0 < K; k0 += 32) {
        FragBF a, b;
#pragma unroll
        for (int r = 0; r < 8; ++r)
            a.u[r] = *(const unsigned*)(arow + k0 + kg + 2 * r);
#pragma unroll
        for (int r = 0; r < 8; ++r)
            b.u[r] = *(const unsigned*)(brow + k0 + 2 * r);
        acc = __builtin_amdgcn_wmma_f32_16x16x32_bf16(false, a.v, false, b.v,
                                                      (short)0, acc, false, false);
    }
    const int   n  = n0 + (lane & 15);
    const float bv = bias[n];
    const int   mb = (lane >> 4) * 8;   // C: lanes 16-31 hold M=8..15
#pragma unroll
    for (int r = 0; r < 8; ++r)
        C[(size_t)(mt * 16 + mb + r) * N + n] = acc[r] + bv;
}

// ---------------------------------------------------------------------------
// Persistent LSTM recurrence.  NWG workgroups, 256 threads (8 waves) each.
// Workgroup wg owns H-columns [wg*32, wg*32+32) for all 4 gates (i,f,g,o),
// i.e. 128 columns of U.  The U slice (128 x 1024 bf16, 258 KB padded) is
// staged ONCE into the 320 KB LDS; h_{t-1} (32 KB) is staged EVERY step via
// CDNA5 async global->LDS loads (ASYNCcnt).  Per step the gate GEMM is
// M=16 x N=128 x K=1024 bf16 WMMA from LDS (32 WMMAs per wave), followed by
// the fused sigmoid/tanh cell update.  h is double-buffered (bf16) in global
// scratch; steps are separated by a release/acquire software grid barrier.
// ---------------------------------------------------------------------------
__global__ __launch_bounds__(256) void lstm_recurrent(
    const float* __restrict__ xproj,        // [ROWS][4H], row = b*T + t
    const unsigned short* __restrict__ Ut,  // [4H][H] bf16 (U transposed)
    float* __restrict__ hs,                 // [ROWS][H] fp32 output states
    unsigned short* __restrict__ hbuf,      // [2][B][H] bf16 (double buffer)
    unsigned* __restrict__ bar) {
    extern __shared__ unsigned short ldsDyn[];
    unsigned short* Us = ldsDyn;            // [128][UPADE]
    unsigned short* hS = ldsDyn + ULDSE;    // [BSZ][UPADE]
    const unsigned hS_off = (unsigned)(uintptr_t)(void*)hS;  // LDS byte offset

    __shared__ float gatesS[4][BSZ][32];
    __shared__ float cS[BSZ][32];

    const int tid   = threadIdx.x;
    const int wave  = tid >> 5;
    const int lane  = tid & 31;
    const int wg    = blockIdx.x;
    const int gw    = wave >> 1;                       // gate id 0..3
    const int ns    = wave & 1;                        // 16-col subtile
    const int hbase = wg * 32;
    const int nloc  = gw * 32 + ns * 16 + (lane & 15); // local U row 0..127
    const int nglob = gw * HID + hbase + ns * 16 + (lane & 15);
    const int mb    = (lane >> 4) * 8;                 // C rows per lane group

    // ---- one-shot: stage this WG's 128 x 1024 bf16 U slice into LDS ----
    for (int e = tid; e < 128 * (HID / 8); e += 256) {   // uint4 granules
        const int row = e >> 7;           // 0..127
        const int kq  = e & 127;          // uint4 index in row
        const int gate = row >> 5, col = row & 31;
        const uint4* src =
            (const uint4*)(Ut + (size_t)(gate * HID + hbase + col) * HID) + kq;
        *(uint4*)(Us + (size_t)row * UPADE + kq * 8) = *src;
    }

    // ---- init: c = 0, own slice of h_{-1} (parity 0) = 0 ----
    for (int e = tid; e < BSZ * 32; e += 256) {
        const int b = e >> 5, j = e & 31;
        cS[b][j] = 0.f;
        hbuf[(size_t)b * HID + hbase + j] = 0;
    }
    grid_barrier(bar, NWG);

    const int kb = (lane < 16) ? 0 : 8;
    const int kg = (lane < 16) ? 0 : 16;
    const unsigned short* hrow = hS + (size_t)(lane & 15) * UPADE;
    const unsigned short* brow = Us + (size_t)nloc * UPADE + kg;

    // async h staging assignment: thread -> (row, 128B chunk)
    const int srow   = tid >> 4;    // 0..15
    const int schunk = tid & 15;    // 0..15
    const unsigned sl0 = hS_off + (unsigned)(srow * (UPADE * 2) + schunk * 128);

    for (int t = 0; t < TLEN; ++t) {
        // ---- stage h_{t-1} into LDS with CDNA5 async global->LDS b128 ----
        {
            const unsigned short* g0 =
                hbuf + (size_t)(t & 1) * BSZ * HID + (size_t)srow * HID + schunk * 64;
#pragma unroll
            for (int i = 0; i < 8; ++i) {
                const unsigned lo = sl0 + (unsigned)(i * 16);
                const unsigned short* ga = g0 + i * 8;
                asm volatile("global_load_async_to_lds_b128 %0, %1, off"
                             :: "v"(lo), "v"(ga) : "memory");
            }
            asm volatile("s_wait_asynccnt 0x0" ::: "memory");
        }
        // prefetch next step's gate pre-activations while we compute
        if (t + 1 < TLEN) {
#pragma unroll
            for (int r = 0; r < 8; ++r)
                __builtin_prefetch(
                    &xproj[((size_t)(mb + r) * TLEN + (t + 1)) * FOURH + nglob], 0, 0);
        }
        __syncthreads();

        // ---- gates = h_{t-1} @ U   (all fragments from LDS) ----
        v8f acc = {};
#pragma unroll 4
        for (int kk = 0; kk < HID / 32; ++kk) {
            const int k0 = kk * 32;
            FragBF a, b;
            a.q[0] = *(const uint4*)(hrow + k0 + kg);
            a.q[1] = *(const uint4*)(hrow + k0 + kg + 8);
            b.q[0] = *(const uint4*)(brow + k0);
            b.q[1] = *(const uint4*)(brow + k0 + 8);
            acc = __builtin_amdgcn_wmma_f32_16x16x32_bf16(false, a.v, false, b.v,
                                                          (short)0, acc, false, false);
        }

        // accumulate precomputed x@W+b and stash gate pre-activations in LDS
#pragma unroll
        for (int r = 0; r < 8; ++r) {
            const int b = mb + r;   // batch index
            const float z =
                acc[r] + xproj[((size_t)b * TLEN + t) * FOURH + nglob];
            gatesS[gw][b][ns * 16 + (lane & 15)] = z;
        }
        __syncthreads();

        // ---- fused elementwise LSTM cell update (512 elems, 2/thread) ----
        unsigned short* hn = hbuf + (size_t)((t + 1) & 1) * BSZ * HID;
        for (int e = tid; e < BSZ * 32; e += 256) {
            const int b = e >> 5, j = e & 31;
            const float ig = sigmoidf_(gatesS[0][b][j]);
            const float fg = sigmoidf_(gatesS[1][b][j]);
            const float gg = tanhf(gatesS[2][b][j]);
            const float og = sigmoidf_(gatesS[3][b][j]);
            const float c  = fg * cS[b][j] + ig * gg;
            cS[b][j] = c;
            const float h = og * tanhf(c);
            hs[((size_t)b * TLEN + t) * HID + hbase + j] = h;
            hn[(size_t)b * HID + hbase + j] = f2bf(h);
        }
        grid_barrier(bar, (unsigned)NWG * (unsigned)(t + 2));
    }
}

// ---------------------------------------------------------------------------
// LayerNorm over H=1024, writing bf16 (feeds the next WMMA GEMM).
// One wave per row; 8 rows per 256-thread block.
// ---------------------------------------------------------------------------
__global__ __launch_bounds__(256) void layernorm_bf16(
    const float* __restrict__ X, const float* __restrict__ g,
    const float* __restrict__ be, unsigned short* __restrict__ Y) {
    const int row  = blockIdx.x * 8 + (threadIdx.x >> 5);
    const int lane = threadIdx.x & 31;
    const float* x = X + (size_t)row * HID;

    float s = 0.f, s2 = 0.f;
    for (int j = lane; j < HID; j += 32) {
        const float v = x[j];
        s += v;
        s2 += v * v;
    }
#pragma unroll
    for (int off = 16; off > 0; off >>= 1) {
        s  += __shfl_xor(s, off);
        s2 += __shfl_xor(s2, off);
    }
    const float mu  = s * (1.0f / HID);
    const float var = s2 * (1.0f / HID) - mu * mu;
    const float rs  = rsqrtf(var + 1e-3f);
    unsigned short* y = Y + (size_t)row * HID;
    for (int j = lane; j < HID; j += 32)
        y[j] = f2bf((x[j] - mu) * rs * g[j] + be[j]);
}

// ---------------------------------------------------------------------------
// Host-side orchestration
// ---------------------------------------------------------------------------
static inline size_t align256(size_t x) { return (x + 255) & ~(size_t)255; }

extern "C" void kernel_launch(void* const* d_in, const int* in_sizes, int n_in,
                              void* d_out, int out_size, void* d_ws, size_t ws_size,
                              hipStream_t stream) {
    (void)in_sizes; (void)n_in; (void)out_size; (void)ws_size;

    const int*   tokens = (const int*)  d_in[0];
    const float* embed  = (const float*)d_in[1];
    const float* W0  = (const float*)d_in[2];
    const float* U0  = (const float*)d_in[3];
    const float* b0  = (const float*)d_in[4];
    const float* g0  = (const float*)d_in[5];
    const float* be0 = (const float*)d_in[6];
    const float* Wd0 = (const float*)d_in[7];
    const float* bd0 = (const float*)d_in[8];
    const float* W1  = (const float*)d_in[9];
    const float* U1  = (const float*)d_in[10];
    const float* b1  = (const float*)d_in[11];
    const float* g1  = (const float*)d_in[12];
    const float* be1 = (const float*)d_in[13];
    const float* Wd1 = (const float*)d_in[14];
    const float* bd1 = (const float*)d_in[15];
    float* out = (float*)d_out;

    // workspace carve-up
    char* ws = (char*)d_ws;
    size_t off = 0;
    auto take = [&](size_t bytes) {
        char* p = ws + off;
        off = align256(off + bytes);
        return p;
    };
    unsigned*       bar   = (unsigned*)      take(1024);                 // 2 counters
    unsigned short* W0T   = (unsigned short*)take((size_t)FOURH * EMB * 2);
    unsigned short* U0T   = (unsigned short*)take((size_t)FOURH * HID * 2);
    unsigned short* Wd0T  = (unsigned short*)take((size_t)PDIM * HID * 2);
    unsigned short* W1T   = (unsigned short*)take((size_t)FOURH * PDIM * 2);
    unsigned short* U1T   = (unsigned short*)take((size_t)FOURH * HID * 2);
    unsigned short* Wd1T  = (unsigned short*)take((size_t)PDIM * HID * 2);
    unsigned short* Xb    = (unsigned short*)take((size_t)ROWS * EMB * 2);
    float*          Xproj = (float*)         take((size_t)ROWS * FOURH * 4);
    float*          Hs    = (float*)         take((size_t)ROWS * HID * 4);
    unsigned short* Hsn   = (unsigned short*)take((size_t)ROWS * HID * 2);
    float*          X1f   = (float*)         take((size_t)ROWS * PDIM * 4);
    unsigned short* X1b   = (unsigned short*)take((size_t)ROWS * PDIM * 2);
    unsigned short* Hbuf  = (unsigned short*)take((size_t)2 * BSZ * HID * 2);

    zero_u32<<<1, 256, 0, stream>>>(bar, 256);

    // weight transpose + bf16 convert (one-shot, deterministic)
    transpose_f32_bf16<<<8192, 256, 0, stream>>>(W0,  W0T,  EMB,  FOURH);
    transpose_f32_bf16<<<16384, 256, 0, stream>>>(U0,  U0T,  HID,  FOURH);
    transpose_f32_bf16<<<2560, 256, 0, stream>>>(Wd0, Wd0T, HID,  PDIM);
    transpose_f32_bf16<<<10240, 256, 0, stream>>>(W1,  W1T,  PDIM, FOURH);
    transpose_f32_bf16<<<16384, 256, 0, stream>>>(U1,  U1T,  HID,  FOURH);
    transpose_f32_bf16<<<2560, 256, 0, stream>>>(Wd1, Wd1T, HID,  PDIM);

    // embedding gather -> bf16
    gather_embed<<<ROWS, 256, 0, stream>>>(tokens, embed, Xb);

    // ---- layer 0 ----
    gemm_bf16_wmma<<<dim3(ROWS / 16, FOURH / 128), 256, 0, stream>>>(
        Xb, W0T, b0, Xproj, ROWS, FOURH, EMB);
    lstm_recurrent<<<NWG, 256, DYNLDS_BYTES, stream>>>(Xproj, U0T, Hs, Hbuf, bar);
    layernorm_bf16<<<ROWS / 8, 256, 0, stream>>>(Hs, g0, be0, Hsn);
    gemm_bf16_wmma<<<dim3(ROWS / 16, PDIM / 128), 256, 0, stream>>>(
        Hsn, Wd0T, bd0, X1f, ROWS, PDIM, HID);
    cvt_f32_bf16<<<8192, 256, 0, stream>>>(X1f, X1b, (size_t)ROWS * PDIM);

    // ---- layer 1 ----
    gemm_bf16_wmma<<<dim3(ROWS / 16, FOURH / 128), 256, 0, stream>>>(
        X1b, W1T, b1, Xproj, ROWS, FOURH, PDIM);
    lstm_recurrent<<<NWG, 256, DYNLDS_BYTES, stream>>>(Xproj, U1T, Hs, Hbuf, bar + 64);
    layernorm_bf16<<<ROWS / 8, 256, 0, stream>>>(Hs, g1, be1, Hsn);
    gemm_bf16_wmma<<<dim3(ROWS / 16, PDIM / 128), 256, 0, stream>>>(
        Hsn, Wd1T, bd1, out, ROWS, PDIM, HID);
}
